// GroupedQueryAttention_71038759075920
// MI455X (gfx1250) — matrix operand order repaired and checked by reading, not compiled, not run on hardware
//
#include <hip/hip_runtime.h>
#include <cstdint>

#define S_LEN   2048
#define D_MODEL 2048
#define NH 32
#define NG 8
#define HDIM 64

typedef __bf16 bf16_t;
typedef bf16_t v16bf __attribute__((ext_vector_type(16)));
typedef float  v8f   __attribute__((ext_vector_type(8)));
typedef int    vsi4  __attribute__((vector_size(4 * sizeof(int))));

union Frag { v16bf v; unsigned u[8]; };

__device__ __forceinline__ unsigned pack_bf16(float a, float b) {
    unsigned ua = __float_as_uint(a), ub = __float_as_uint(b);
    ua += 0x7FFFu + ((ua >> 16) & 1u);   // RNE
    ub += 0x7FFFu + ((ub >> 16) & 1u);
    return (ua >> 16) | (ub & 0xFFFF0000u);
}
__device__ __forceinline__ unsigned short bf16_1(float a) {
    unsigned ua = __float_as_uint(a);
    ua += 0x7FFFu + ((ua >> 16) & 1u);
    return (unsigned short)(ua >> 16);
}

// ---- CDNA5 async global->LDS (ASYNCcnt) ------------------------------------
__device__ __forceinline__ void async_copy_b128(const void* g, void* l) {
#if __has_builtin(__builtin_amdgcn_global_load_async_to_lds_b128)
    __builtin_amdgcn_global_load_async_to_lds_b128(
        (__attribute__((address_space(1))) vsi4*)(void*)g,
        (__attribute__((address_space(3))) vsi4*)l, 0, 0);
#else
    unsigned loff = (unsigned)(unsigned long long)
        (__attribute__((address_space(3))) char*)l;
    asm volatile("global_load_async_to_lds_b128 %0, %1, off"
                 :: "v"(loff), "v"(g) : "memory");
#endif
}
__device__ __forceinline__ void wait_async0() {
#if __has_builtin(__builtin_amdgcn_s_wait_asynccnt)
    __builtin_amdgcn_s_wait_asynccnt(0);
#else
    asm volatile("s_wait_asynccnt 0x0" ::: "memory");
#endif
}

// ---------------------------------------------------------------------------
// fp32 -> bf16 one-time conversion (2 elems/thread, packed dword stores)
// ---------------------------------------------------------------------------
__global__ __launch_bounds__(256) void f32_to_bf16_kernel(
    const float* __restrict__ s, unsigned short* __restrict__ d, int n)
{
    int i = (blockIdx.x * 256 + threadIdx.x) * 2;
    if (i < n) *(unsigned*)(d + i) = pack_bf16(s[i], s[i + 1]);
}

// ---------------------------------------------------------------------------
// GEMM: C[MxN] = A[MxK] * B[NxK]^T, A/B bf16 row-major, C fp32.
// 128 threads (4 waves), block tile 128x128, wave tile 64x64 (4x4 WMMA).
// Double-buffered async-to-LDS staging.
// ---------------------------------------------------------------------------
#define LDW 20   // uints per LDS row (16 data + 4 pad) -> 80B, 16B aligned

__global__ __launch_bounds__(128) void gemm_bt_bf16_kernel(
    const unsigned short* __restrict__ A, const unsigned short* __restrict__ B,
    float* __restrict__ C, int M, int N, int K)
{
    __shared__ unsigned sA[2][128 * LDW];
    __shared__ unsigned sB[2][128 * LDW];

    const int tid = threadIdx.x;
    const int lane = tid & 31, wave = tid >> 5;
    const int lq = lane & 15, lh = lane >> 4;
    const int bM = blockIdx.y * 128, bN = blockIdx.x * 128;
    const int wM = (wave >> 1) * 64, wN = (wave & 1) * 64;

    v8f acc[4][4];
#pragma unroll
    for (int mt = 0; mt < 4; ++mt)
#pragma unroll
        for (int nt = 0; nt < 4; ++nt)
#pragma unroll
            for (int i = 0; i < 8; ++i) acc[mt][nt][i] = 0.f;

    const unsigned short* ga = A + (size_t)(bM + tid) * K;
    const unsigned short* gb = B + (size_t)(bN + tid) * K;

    auto stage = [&](int kc, int buf) {
        unsigned* da = &sA[buf][tid * LDW];
        unsigned* db = &sB[buf][tid * LDW];
#pragma unroll
        for (int j = 0; j < 4; ++j) {
            async_copy_b128(ga + kc + j * 8, da + j * 4);
            async_copy_b128(gb + kc + j * 8, db + j * 4);
        }
    };

    const int nk = K >> 5;
    stage(0, 0);
    for (int ic = 0; ic < nk; ++ic) {
        wait_async0();
        __syncthreads();
        if (ic + 1 < nk) stage((ic + 1) << 5, (ic + 1) & 1);

        const unsigned* bufA = sA[ic & 1];
        const unsigned* bufB = sB[ic & 1];
        Frag aF[4], bF[4];
#pragma unroll
        for (int mt = 0; mt < 4; ++mt) {
            const unsigned* p = &bufA[(wM + mt * 16 + lq) * LDW];
#pragma unroll
            for (int v = 0; v < 8; ++v)
                aF[mt].u[v] = p[v + (lh << 3)];
        }
#pragma unroll
        for (int nt = 0; nt < 4; ++nt) {
            const unsigned* p = &bufB[(wN + nt * 16 + lq) * LDW];
#pragma unroll
            for (int v = 0; v < 8; ++v)
                bF[nt].u[v] = p[v + (lh << 3)];
        }
#pragma unroll
        for (int mt = 0; mt < 4; ++mt)
#pragma unroll
            for (int nt = 0; nt < 4; ++nt)
                acc[mt][nt] = __builtin_amdgcn_wmma_f32_16x16x32_bf16(
                    false, aF[mt].v, false, bF[nt].v, (short)0, acc[mt][nt], false, false);
    }

#pragma unroll
    for (int mt = 0; mt < 4; ++mt)
#pragma unroll
        for (int nt = 0; nt < 4; ++nt) {
            const int row = bM + wM + mt * 16 + lh * 8;
            const int col = bN + wN + nt * 16 + lq;
            float* cp = C + (size_t)row * N + col;
#pragma unroll
            for (int v = 0; v < 8; ++v)
                cp[(size_t)v * N] = acc[mt][nt][v];
        }
}

// ---------------------------------------------------------------------------
// RMSNorm + RoPE, wave per (s,head) row; emits bf16 in attention layouts:
// q -> [H][S][64], k -> [G][S][64], v -> [G][S/64][64 hd][64 key] (block-T)
// ---------------------------------------------------------------------------
__global__ __launch_bounds__(256) void rope_norm_kernel(
    const float* __restrict__ qg, const float* __restrict__ kw, const float* __restrict__ vw,
    const float* __restrict__ cosb, const float* __restrict__ sinb,
    const float* __restrict__ qnw, const float* __restrict__ knw,
    unsigned short* __restrict__ qr, unsigned short* __restrict__ kr,
    unsigned short* __restrict__ vt)
{
    const int QROWS = S_LEN * NH;
    const int KROWS = S_LEN * NG;
    const int wid = blockIdx.x * 8 + (threadIdx.x >> 5);
    const int lane = threadIdx.x & 31;

    if (wid < QROWS + KROWS) {
        const bool isQ = wid < QROWS;
        int s; const float* in; const float* nw; unsigned short* out;
        if (isQ) {
            s = wid / NH; const int h = wid % NH;
            in = qg + (size_t)s * (NH * 2 * HDIM) + h * (2 * HDIM);
            nw = qnw;
            out = qr + ((size_t)h * S_LEN + s) * HDIM;
        } else {
            const int rk = wid - QROWS;
            s = rk / NG; const int g = rk % NG;
            in = kw + (size_t)s * (NG * HDIM) + g * HDIM;
            nw = knw;
            out = kr + ((size_t)g * S_LEN + s) * HDIM;
        }
        float x1 = in[lane], x2 = in[lane + 32];
        float ss = x1 * x1 + x2 * x2;
#pragma unroll
        for (int m = 1; m < 32; m <<= 1) ss += __shfl_xor(ss, m, 32);
        float r = rsqrtf(ss * (1.0f / 64.0f) + 1e-6f);
        float n1 = x1 * r * (1.f + nw[lane]);
        float n2 = x2 * r * (1.f + nw[lane + 32]);
        float c1 = cosb[s * 64 + lane], c2 = cosb[s * 64 + lane + 32];
        float s1 = sinb[s * 64 + lane], s2 = sinb[s * 64 + lane + 32];
        out[lane]      = bf16_1(n1 * c1 - n2 * s1);
        out[lane + 32] = bf16_1(n2 * c2 + n1 * s2);
    } else if (wid < QROWS + 2 * KROWS) {
        const int rv = wid - QROWS - KROWS;
        const int s = rv / NG, g = rv % NG;
        const float* in = vw + (size_t)s * (NG * HDIM) + g * HDIM;
        const int kb = s >> 6, kwi = s & 63;
        unsigned short* outb = vt + (((size_t)g * (S_LEN / 64) + kb) * 64) * 64 + kwi;
        outb[(size_t)lane * 64]        = bf16_1(in[lane]);
        outb[(size_t)(lane + 32) * 64] = bf16_1(in[lane + 32]);
    }
}

// ---------------------------------------------------------------------------
// Flash attention, transposed scores (S^T = K*Q^T) -> lane-local softmax,
// P^T packs directly into the P*V A-fragment. Async double-buffered K/V tiles.
// ---------------------------------------------------------------------------
#define LKV 36   // uints per K/V LDS row (32 data + 4 pad) -> 144B, 16B aligned

__global__ __launch_bounds__(128) void attn_kernel(
    const unsigned short* __restrict__ Q,   // [H][S][64] bf16
    const unsigned short* __restrict__ Kh,  // [G][S][64] bf16
    const unsigned short* __restrict__ Vt,  // [G][S/64][64][64] bf16 (block-T)
    float* __restrict__ O)                  // [H][S][64] fp32
{
    __shared__ unsigned sK[2][64 * LKV];
    __shared__ unsigned sV[2][64 * LKV];

    const int tid = threadIdx.x;
    const int lane = tid & 31;
    const int wave = tid >> 5;
    const int lq = lane & 15, lh = lane >> 4;
    const int h = blockIdx.y;
    const int g = h >> 2;                   // GS = H/G = 4
    const int qb = blockIdx.x * 64;
    const int qrow = qb + wave * 16 + lq;

    Frag bQ[2];                             // loop-invariant Q^T B-fragments
    {
        const unsigned short* qp = Q + ((size_t)h * S_LEN + qb + wave * 16 + lq) * HDIM;
#pragma unroll
        for (int c = 0; c < 2; ++c)
#pragma unroll
            for (int v = 0; v < 8; ++v)
                bQ[c].u[v] = *(const unsigned*)(qp + c * 32 + lh * 16 + 2 * v);
    }

    const int row2 = tid >> 1, half = tid & 1;
    auto stage = [&](int kb, int buf) {
        const unsigned short* kp = Kh + ((size_t)g * S_LEN + kb * 64 + row2) * HDIM + half * 32;
        const unsigned short* vp = Vt + (((size_t)g * (S_LEN / 64) + kb) * 64 + row2) * 64 + half * 32;
        unsigned* dk = &sK[buf][row2 * LKV + half * 16];
        unsigned* dv = &sV[buf][row2 * LKV + half * 16];
#pragma unroll
        for (int j = 0; j < 4; ++j) {
            async_copy_b128(kp + j * 8, dk + j * 4);
            async_copy_b128(vp + j * 8, dv + j * 4);
        }
    };

    v8f acc[4];
#pragma unroll
    for (int nt = 0; nt < 4; ++nt)
#pragma unroll
        for (int i = 0; i < 8; ++i) acc[nt][i] = 0.f;
    float mrun = -3.0e38f, ssum = 0.f;

    const int nkb = blockIdx.x + 1;         // causal
    stage(0, 0);
    for (int kb = 0; kb < nkb; ++kb) {
        wait_async0();
        __syncthreads();
        if (kb + 1 < nkb) stage(kb + 1, (kb + 1) & 1);
        const unsigned* bK = sK[kb & 1];
        const unsigned* bV_ = sV[kb & 1];
        const int kbase = kb * 64;

        float p[4][8];
        float bm = -3.0e38f;
#pragma unroll
        for (int mt = 0; mt < 4; ++mt) {    // S^T tile: rows=keys, cols=queries
            v8f s;
#pragma unroll
            for (int i = 0; i < 8; ++i) s[i] = 0.f;
#pragma unroll
            for (int c = 0; c < 2; ++c) {
                Frag aK;
                const unsigned* pk = &bK[(mt * 16 + lq) * LKV + c * 16];
#pragma unroll
                for (int v = 0; v < 8; ++v)
                    aK.u[v] = pk[v + (lh << 3)];
                s = __builtin_amdgcn_wmma_f32_16x16x32_bf16(
                    false, aK.v, false, bQ[c].v, (short)0, s, false, false);
            }
#pragma unroll
            for (int v = 0; v < 8; ++v) {
                int key = kbase + mt * 16 + lh * 8 + v;
                float val = (key <= qrow) ? s[v] * 0.125f : -3.0e38f;
                p[mt][v] = val;
                bm = fmaxf(bm, val);
            }
        }
        bm = fmaxf(bm, __shfl_xor(bm, 16, 32));
        float mnew = fmaxf(mrun, bm);
        float corr = __expf(mrun - mnew);
        float bs = 0.f;
#pragma unroll
        for (int mt = 0; mt < 4; ++mt)
#pragma unroll
            for (int v = 0; v < 8; ++v) {
                float e = __expf(p[mt][v] - mnew);
                p[mt][v] = e;
                bs += e;
            }
        bs += __shfl_xor(bs, 16, 32);
        ssum = ssum * corr + bs;
        mrun = mnew;

        float f[8];
#pragma unroll
        for (int v = 0; v < 8; ++v) f[v] = __shfl(corr, v + 8 * lh, 32);
#pragma unroll
        for (int nt = 0; nt < 4; ++nt)
#pragma unroll
            for (int v = 0; v < 8; ++v) acc[nt][v] *= f[v];

        Frag pA[2];                          // P^T C-layout -> P A-fragment
#pragma unroll
        for (int c = 0; c < 2; ++c)
#pragma unroll
            for (int v = 0; v < 8; ++v) {
                int t = c * 2 + (v >> 2);
                int e = (v & 3) * 2;
                pA[c].u[v] = pack_bf16(p[t][e], p[t][e + 1]);
            }
#pragma unroll
        for (int nt = 0; nt < 4; ++nt)
#pragma unroll
            for (int c = 0; c < 2; ++c) {
                Frag bV;
                const unsigned* pv = &bV_[(nt * 16 + lq) * LKV + c * 16];
#pragma unroll
                for (int v = 0; v < 8; ++v) bV.u[v] = pv[(v & 3) + ((v & 4) << 1) + (lh << 2)];
                acc[nt] = __builtin_amdgcn_wmma_f32_16x16x32_bf16(
                    false, pA[c].v, false, bV.v, (short)0, acc[nt], false, false);
            }
    }

    float inv = 1.f / ssum;
    float f[8];
#pragma unroll
    for (int v = 0; v < 8; ++v) f[v] = __shfl(inv, v + 8 * lh, 32);
#pragma unroll
    for (int nt = 0; nt < 4; ++nt)
#pragma unroll
        for (int v = 0; v < 8; ++v)
            O[((size_t)h * S_LEN + qb + wave * 16 + v + 8 * lh) * HDIM + nt * 16 + lq] =
                acc[nt][v] * f[v];
}

// ---------------------------------------------------------------------------
// ctx * sigmoid(gate) -> bf16 [s][h*64+j] for the output projection.
// ---------------------------------------------------------------------------
__global__ __launch_bounds__(256) void gate_kernel(
    const float* __restrict__ ctx, const float* __restrict__ qg,
    unsigned short* __restrict__ outg)
{
    int i2 = (blockIdx.x * 256 + threadIdx.x) * 2;
    if (i2 >= S_LEN * D_MODEL) return;
    int s = i2 >> 11, c = i2 & 2047;
    int h = c >> 6, j = c & 63;
    const float* gp = qg + (size_t)s * (NH * 2 * HDIM) + h * (2 * HDIM) + HDIM + j;
    const float* cp = ctx + ((size_t)h * S_LEN + s) * HDIM + j;
    float a = cp[0] / (1.f + __expf(-gp[0]));
    float b = cp[1] / (1.f + __expf(-gp[1]));
    *(unsigned*)(outg + i2) = pack_bf16(a, b);
}

// ---------------------------------------------------------------------------
extern "C" void kernel_launch(void* const* d_in, const int* in_sizes, int n_in,
                              void* d_out, int out_size, void* d_ws, size_t ws_size,
                              hipStream_t stream) {
    const float* x    = (const float*)d_in[0];
    // d_in[1] = mask (bool) -- causality computed analytically
    const float* cosb = (const float*)d_in[2];
    const float* sinb = (const float*)d_in[3];
    const float* Wq   = (const float*)d_in[4];
    const float* Wk   = (const float*)d_in[5];
    const float* Wv   = (const float*)d_in[6];
    const float* Wo   = (const float*)d_in[7];
    const float* qnw  = (const float*)d_in[8];
    const float* knw  = (const float*)d_in[9];

    float* fw = (float*)d_ws;
    size_t fo = 0;
    float* qg  = fw + fo; fo += (size_t)S_LEN * 4096;   // QG (q | gate) fp32
    float* kw  = fw + fo; fo += (size_t)S_LEN * 512;
    float* vw  = fw + fo; fo += (size_t)S_LEN * 512;
    float* ctx = fw + fo; fo += (size_t)S_LEN * 2048;   // [H][S][64] fp32

    unsigned short* uw = (unsigned short*)(fw + fo);
    size_t uo = 0;
    unsigned short* xb  = uw + uo; uo += (size_t)S_LEN * 2048;
    unsigned short* wqb = uw + uo; uo += (size_t)4096 * 2048;
    unsigned short* wkb = uw + uo; uo += (size_t)512 * 2048;
    unsigned short* wvb = uw + uo; uo += (size_t)512 * 2048;
    unsigned short* wob = uw + uo; uo += (size_t)2048 * 2048;
    unsigned short* qr  = uw + uo; uo += (size_t)S_LEN * 2048;  // [H][S][64]
    unsigned short* kr  = uw + uo; uo += (size_t)S_LEN * 512;   // [G][S][64]
    unsigned short* vt  = uw + uo; uo += (size_t)S_LEN * 512;   // [G][32][64][64]
    unsigned short* gtd = uw + uo; uo += (size_t)S_LEN * 2048;  // [S][2048]

    // one-time fp32 -> bf16 conversions
    f32_to_bf16_kernel<<<(S_LEN * 2048 / 2 + 255) / 256, 256, 0, stream>>>(x, xb, S_LEN * 2048);
    f32_to_bf16_kernel<<<(4096 * 2048 / 2 + 255) / 256, 256, 0, stream>>>(Wq, wqb, 4096 * 2048);
    f32_to_bf16_kernel<<<(512 * 2048 / 2 + 255) / 256, 256, 0, stream>>>(Wk, wkb, 512 * 2048);
    f32_to_bf16_kernel<<<(512 * 2048 / 2 + 255) / 256, 256, 0, stream>>>(Wv, wvb, 512 * 2048);
    f32_to_bf16_kernel<<<(2048 * 2048 / 2 + 255) / 256, 256, 0, stream>>>(Wo, wob, 2048 * 2048);

    // QKV projections
    gemm_bt_bf16_kernel<<<dim3(4096 / 128, S_LEN / 128), 128, 0, stream>>>(xb, wqb, qg, S_LEN, 4096, D_MODEL);
    gemm_bt_bf16_kernel<<<dim3(512 / 128,  S_LEN / 128), 128, 0, stream>>>(xb, wkb, kw, S_LEN, 512, D_MODEL);
    gemm_bt_bf16_kernel<<<dim3(512 / 128,  S_LEN / 128), 128, 0, stream>>>(xb, wvb, vw, S_LEN, 512, D_MODEL);

    // RMSNorm + RoPE + bf16 relayout
    rope_norm_kernel<<<(S_LEN * NH + 2 * S_LEN * NG) / (8 * 32), 256, 0, stream>>>(
        qg, kw, vw, cosb, sinb, qnw, knw, qr, kr, vt);

    // Flash attention
    attn_kernel<<<dim3(S_LEN / 64, NH), 128, 0, stream>>>(qr, kr, vt, ctx);

    // Gating -> bf16
    gate_kernel<<<(S_LEN * D_MODEL / 2 + 255) / 256, 256, 0, stream>>>(ctx, qg, gtd);

    // Output projection
    gemm_bt_bf16_kernel<<<dim3(D_MODEL / 128, S_LEN / 128), 128, 0, stream>>>(
        gtd, wob, (float*)d_out, S_LEN, D_MODEL, D_MODEL);
}
